// GCNsNet_26027501814346
// MI455X (gfx1250) — matrix operand order repaired and checked by reading, not compiled, not run on hardware
//
#include <hip/hip_runtime.h>
#include <math.h>

typedef __attribute__((ext_vector_type(8)))  float  v8f;
typedef __attribute__((ext_vector_type(16))) __bf16 v16bf;

#define NGRAPH 2048
#define EPG    256                  // edges per graph (32 nodes * degree 8)
#define NEDGE  (NGRAPH * EPG)
#define BN_EPS 1e-5f

__device__ __forceinline__ float softplus_f(float x) {
  return (x > 20.f) ? x : log1pf(expf(x));
}

// ---- transpose + f32->bf16 weight prep: W[k][n] -> Wt[n][k] ----------------
__global__ void prep_weights_kernel(const float* __restrict__ W,
                                    __bf16* __restrict__ Wt, int K, int N) {
  int idx = blockIdx.x * blockDim.x + threadIdx.x;
  if (idx >= K * N) return;
  int n = idx / K, k = idx - n * K;
  Wt[idx] = (__bf16)W[(size_t)k * N + n];
}

// ---- fused per-graph ChebConv (K=2): h = x@Wa + T1x@Wb + bias --------------
// One block = one graph. Edge scatter via LDS atomics, GEMM via bf16 WMMA.
template<int NODES, int NPAD, int CIN, int COUT, int LVL>
__global__ __launch_bounds__(256)
void conv_kernel(const float* __restrict__ xin,
                 const int*   __restrict__ esrc, const int* __restrict__ edst,
                 const float* __restrict__ ewt,
                 const __bf16* __restrict__ wta, const __bf16* __restrict__ wtb,
                 const float* __restrict__ bias,
                 float* __restrict__ hout) {
  __shared__ float  xs [NPAD * CIN];
  __shared__ float  txs[NPAD * CIN];
  __shared__ __bf16 xsb[NPAD * CIN];
  __shared__ __bf16 txb[NPAD * CIN];
  __shared__ float  deg [NPAD];
  __shared__ float  dinv[NPAD];
  __shared__ int    slA[EPG];
  __shared__ int    dlA[EPG];
  __shared__ float  nrm[EPG];

  const int tid = threadIdx.x;
  const int g   = blockIdx.x;

  // load activations (pad rows to NPAD with zeros), zero scatter target
  for (int i = tid; i < NPAD * CIN; i += 256) {
    int r = i / CIN;
    xs[i]  = (r < NODES) ? xin[(size_t)g * NODES * CIN + i] : 0.f;
    txs[i] = 0.f;
  }
  if (tid < NPAD) deg[tid] = 0.f;
  __syncthreads();

  // one thread <-> one edge of this graph; pooled index = original >> LVL,
  // pooled self-loops get zero weight (cumulative across pool levels)
  const int   e   = g * EPG + tid;
  const int   sl  = (esrc[e] >> LVL) - g * NODES;
  const int   dl  = (edst[e] >> LVL) - g * NODES;
  const float w   = ewt[e];
  const float eff = (LVL > 0 && sl == dl) ? 0.f : w;
  atomicAdd(&deg[sl], eff);
  slA[tid] = sl; dlA[tid] = dl;
  __syncthreads();

  if (tid < NPAD) { float d = deg[tid]; dinv[tid] = (d > 0.f) ? rsqrtf(d) : 0.f; }
  __syncthreads();
  nrm[tid] = -dinv[sl] * eff * dinv[dl];
  __syncthreads();

  // scatter:  txs[dst][c] += norm * xs[src][c]  over (edge, channel) pairs
  for (int i = tid; i < EPG * CIN; i += 256) {
    int ee = i / CIN, c = i - ee * CIN;
    atomicAdd(&txs[dlA[ee] * CIN + c], nrm[ee] * xs[slA[ee] * CIN + c]);
  }
  __syncthreads();

  for (int i = tid; i < NPAD * CIN; i += 256) {
    xsb[i] = (__bf16)xs[i];
    txb[i] = (__bf16)txs[i];
  }
  __syncthreads();

  // ---- WMMA GEMM: 8 waves <-> 8 (16x16) output tiles -----------------------
  const int TN   = COUT / 16;
  const int TM   = NPAD / 16;          // TM*TN == 8 for all three layers
  const int wave = tid >> 5;
  const int lane = tid & 31;
  if (wave < TM * TN) {
    const int mt = wave / TN, nt = wave - mt * TN;
    const int half = lane >> 4;        // 0: lanes 0-15, 1: lanes 16-31
    const int mn   = lane & 15;        // A row / B,C,D column within tile
    v8f acc = {};
    for (int k0 = 0; k0 < CIN; k0 += 32) {
      v16bf a, a2, b, b2;
      const int rowoff = (mt * 16 + mn) * CIN;
      // A fragment layout (16-bit A 16x32): VGPR r<4 -> K = half*8 + 2r,
      // r>=4 -> K = 16 + half*8 + 2(r-4); pairs are K-consecutive.
#pragma unroll
      for (int r = 0; r < 8; ++r) {
        int kb = k0 + ((r < 4) ? (half * 8 + 2 * r) : (16 + half * 8 + 2 * (r - 4)));
        a [2 * r]     = xsb[rowoff + kb];
        a [2 * r + 1] = xsb[rowoff + kb + 1];
        a2[2 * r]     = txb[rowoff + kb];
        a2[2 * r + 1] = txb[rowoff + kb + 1];
      }
      // B fragment (32x16): lane column N = mn, K same as A fragment (pairs {2r, 2r+1});
      // weights stored transposed [n][k] so pairs are contiguous bf16.
      const int ncol = nt * 16 + mn;
#pragma unroll
      for (int r = 0; r < 8; ++r) {
        int k = k0 + ((r < 4) ? (half * 8 + 2 * r) : (16 + half * 8 + 2 * (r - 4)));
        b [2 * r]     = wta[(size_t)ncol * CIN + k];
        b [2 * r + 1] = wta[(size_t)ncol * CIN + k + 1];
        b2[2 * r]     = wtb[(size_t)ncol * CIN + k];
        b2[2 * r + 1] = wtb[(size_t)ncol * CIN + k + 1];
      }
      acc = __builtin_amdgcn_wmma_f32_16x16x32_bf16(false, a,  false, b,  (short)0, acc, false, false);
      acc = __builtin_amdgcn_wmma_f32_16x16x32_bf16(false, a2, false, b2, (short)0, acc, false, false);
    }
    // C/D layout: VGPR r -> M = r + 8*half, N = mn
    const float bv = bias[nt * 16 + mn];
#pragma unroll
    for (int r = 0; r < 8; ++r) {
      int row = mt * 16 + r + 8 * half;
      if (row < NODES)
        hout[(size_t)g * NODES * COUT + (size_t)row * COUT + nt * 16 + mn] = acc[r] + bv;
    }
  }
}

// ---- BN statistics: one block per channel, deterministic tree reduce -------
__global__ void stats_kernel(const float* __restrict__ h,
                             float* __restrict__ mean, float* __restrict__ istd,
                             int R, int C) {
  const int c = blockIdx.x, tid = threadIdx.x;
  float s = 0.f, q = 0.f;
  for (int r = tid; r < R; r += 256) {
    float v = h[(size_t)r * C + c];
    s += v; q += v * v;
  }
  __shared__ float ss[256], qq[256];
  ss[tid] = s; qq[tid] = q;
  __syncthreads();
  for (int w = 128; w > 0; w >>= 1) {
    if (tid < w) { ss[tid] += ss[tid + w]; qq[tid] += qq[tid + w]; }
    __syncthreads();
  }
  if (tid == 0) {
    float m = ss[0] / (float)R;
    float v = qq[0] / (float)R - m * m;
    mean[c] = m;
    istd[c] = rsqrtf(v + BN_EPS);
  }
}

// ---- BN + softplus + pair-max pool (softplus monotone => commutes w/ max) --
__global__ void bnpool_kernel(const float* __restrict__ h,
                              const float* __restrict__ mean, const float* __restrict__ istd,
                              const float* __restrict__ gamma, const float* __restrict__ beta,
                              float* __restrict__ hp, int Rout, int C) {
  int idx = blockIdx.x * blockDim.x + threadIdx.x;
  if (idx >= Rout * C) return;
  int r = idx / C, c = idx - r * C;
  float m = mean[c], is = istd[c], gm = gamma[c], bt = beta[c];
  float v0 = softplus_f((h[(size_t)(2 * r)     * C + c] - m) * is * gm + bt);
  float v1 = softplus_f((h[(size_t)(2 * r + 1) * C + c] - m) * is * gm + bt);
  hp[idx] = fmaxf(v0, v1);
}

// ---- BN3 + softplus + graph mean (8 nodes) + FC(128->4) + log_softmax ------
__global__ void final_kernel(const float* __restrict__ h3,
                             const float* __restrict__ mean, const float* __restrict__ istd,
                             const float* __restrict__ gamma, const float* __restrict__ beta,
                             const float* __restrict__ fcW, const float* __restrict__ fcb,
                             float* __restrict__ out) {
  const int g = blockIdx.x, tid = threadIdx.x;   // 128 threads, one per channel
  __shared__ float hg[128];
  __shared__ float lg[4];
  float m = mean[tid], is = istd[tid], gm = gamma[tid], bt = beta[tid];
  float acc = 0.f;
#pragma unroll
  for (int r = 0; r < 8; ++r) {
    float v = h3[((size_t)g * 8 + r) * 128 + tid];
    acc += softplus_f((v - m) * is * gm + bt);
  }
  hg[tid] = acc * 0.125f;
  __syncthreads();
  if (tid < 4) {
    float l = fcb[tid];
    for (int c = 0; c < 128; ++c) l += hg[c] * fcW[c * 4 + tid];
    lg[tid] = l;
  }
  __syncthreads();
  if (tid < 4) {
    float mx  = fmaxf(fmaxf(lg[0], lg[1]), fmaxf(lg[2], lg[3]));
    float sum = expf(lg[0] - mx) + expf(lg[1] - mx) + expf(lg[2] - mx) + expf(lg[3] - mx);
    out[g * 4 + tid] = lg[tid] - mx - logf(sum);
  }
}

extern "C" void kernel_launch(void* const* d_in, const int* in_sizes, int n_in,
                              void* d_out, int out_size, void* d_ws, size_t ws_size,
                              hipStream_t stream) {
  const float* x    = (const float*)d_in[0];
  const float* ewt  = (const float*)d_in[1];
  const float* W1a  = (const float*)d_in[2];
  const float* W1b  = (const float*)d_in[3];
  const float* b1   = (const float*)d_in[4];
  const float* g1   = (const float*)d_in[5];
  const float* be1  = (const float*)d_in[6];
  const float* W2a  = (const float*)d_in[7];
  const float* W2b  = (const float*)d_in[8];
  const float* b2   = (const float*)d_in[9];
  const float* g2   = (const float*)d_in[10];
  const float* be2  = (const float*)d_in[11];
  const float* W3a  = (const float*)d_in[12];
  const float* W3b  = (const float*)d_in[13];
  const float* b3   = (const float*)d_in[14];
  const float* g3   = (const float*)d_in[15];
  const float* be3  = (const float*)d_in[16];
  const float* fcW  = (const float*)d_in[17];
  const float* fcb  = (const float*)d_in[18];
  const int*   eidx = (const int*)d_in[19];
  const int* esrc = eidx;
  const int* edst = eidx + NEDGE;
  float* out = (float*)d_out;
  char*  ws  = (char*)d_ws;

  // workspace layout (bytes)
  __bf16* wt1a = (__bf16*)(ws + 0);
  __bf16* wt1b = (__bf16*)(ws + 8192);
  __bf16* wt2a = (__bf16*)(ws + 16384);
  __bf16* wt2b = (__bf16*)(ws + 32768);
  __bf16* wt3a = (__bf16*)(ws + 49152);
  __bf16* wt3b = (__bf16*)(ws + 81920);
  float* mean1 = (float*)(ws + 114688);
  float* istd1 = mean1 + 64;
  float* mean2 = istd1 + 64;
  float* istd2 = mean2 + 128;
  float* mean3 = istd2 + 128;
  float* istd3 = mean3 + 128;
  float* buf0  = (float*)(ws + (size_t)(128 << 10));                       // 16 MB
  float* buf1  = (float*)(ws + (size_t)(128 << 10) + ((size_t)16 << 20));  //  8 MB

  prep_weights_kernel<<<(64 * 64   + 255) / 256, 256, 0, stream>>>(W1a, wt1a, 64, 64);
  prep_weights_kernel<<<(64 * 64   + 255) / 256, 256, 0, stream>>>(W1b, wt1b, 64, 64);
  prep_weights_kernel<<<(64 * 128  + 255) / 256, 256, 0, stream>>>(W2a, wt2a, 64, 128);
  prep_weights_kernel<<<(64 * 128  + 255) / 256, 256, 0, stream>>>(W2b, wt2b, 64, 128);
  prep_weights_kernel<<<(128 * 128 + 255) / 256, 256, 0, stream>>>(W3a, wt3a, 128, 128);
  prep_weights_kernel<<<(128 * 128 + 255) / 256, 256, 0, stream>>>(W3b, wt3b, 128, 128);

  // layer 1: 65536 x 64 -> 64
  float* h1 = buf0;
  conv_kernel<32, 32, 64, 64, 0><<<NGRAPH, 256, 0, stream>>>(x, esrc, edst, ewt, wt1a, wt1b, b1, h1);
  stats_kernel<<<64, 256, 0, stream>>>(h1, mean1, istd1, 65536, 64);
  float* h1p = buf1;                                   // 32768 x 64
  bnpool_kernel<<<(32768 * 64) / 256, 256, 0, stream>>>(h1, mean1, istd1, g1, be1, h1p, 32768, 64);

  // layer 2: 32768 x 64 -> 128
  float* h2 = buf0;
  conv_kernel<16, 16, 64, 128, 1><<<NGRAPH, 256, 0, stream>>>(h1p, esrc, edst, ewt, wt2a, wt2b, b2, h2);
  stats_kernel<<<128, 256, 0, stream>>>(h2, mean2, istd2, 32768, 128);
  float* h2p = buf1;                                   // 16384 x 128
  bnpool_kernel<<<(16384 * 128) / 256, 256, 0, stream>>>(h2, mean2, istd2, g2, be2, h2p, 16384, 128);

  // layer 3: 16384 x 128 -> 128 (8 real rows per graph, padded to 16 in LDS)
  float* h3 = buf0;
  conv_kernel<8, 16, 128, 128, 2><<<NGRAPH, 256, 0, stream>>>(h2p, esrc, edst, ewt, wt3a, wt3b, b3, h3);
  stats_kernel<<<128, 256, 0, stream>>>(h3, mean3, istd3, 16384, 128);

  final_kernel<<<NGRAPH, 128, 0, stream>>>(h3, mean3, istd3, g3, be3, fcW, fcb, out);

  (void)in_sizes; (void)n_in; (void)out_size; (void)ws_size;
}
